// Attention_35716948034477
// MI455X (gfx1250) — matrix operand order repaired and checked by reading, not compiled, not run on hardware
//
#include <hip/hip_runtime.h>
#include <hip/hip_bf16.h>
#include <math.h>

typedef _Float16 v16h __attribute__((ext_vector_type(16)));
typedef float    v8f  __attribute__((ext_vector_type(8)));

#define HEAD_DIM   128
#define N_HEADS    32
#define N_KV_HEADS 8
#define REPEATS    4
#define DIM        4096
#define START_POS  8191
#define KV_LEN     8192
#define NS         32          // split-K factor: 8192/256 keys
#define CHUNK      256         // keys per block
#define SUB        128         // keys per LDS phase

#define SCALE 0.088388347648318447f   // 1/sqrt(128)

// workspace layout (float offsets)
#define WS_XQ   0              // 4096  raw xq (pre-RoPE)
#define WS_XK   4096           // 1024  raw xk
#define WS_XV   5120           // 1024  raw xv
#define WS_M    6144           // 32*NS per-(qhead,split) max
#define WS_L    7168           // 32*NS per-(qhead,split) sumexp
#define WS_ATTN 8192           // 4096  merged attention output
#define WS_ACC  12288          // 32*NS*128 split partial P.V

// CDNA5 async copy global->LDS (ASYNCcnt path), CDNA5 ISA
__device__ __forceinline__ void async_ld_b128(unsigned lds_byte_addr, const float* gptr) {
    asm volatile("global_load_async_to_lds_b128 %0, %1, off"
                 :: "v"(lds_byte_addr), "v"(gptr) : "memory");
}
#define WAIT_ASYNC(n) asm volatile("s_wait_asynccnt " #n ::: "memory")

// ---------------- QKV projection: one wave per output row ----------------
__global__ __launch_bounds__(256) void qkv_gemv(const float* __restrict__ x,
                                                const float* __restrict__ wq,
                                                const float* __restrict__ wk,
                                                const float* __restrict__ wv,
                                                float* __restrict__ ws) {
    const int wave = threadIdx.x >> 5, lane = threadIdx.x & 31;
    const int r = blockIdx.x * 8 + wave;          // 0..6143
    const float* W;
    float* out;
    if (r < 4096)       { W = wq + (size_t)r * DIM;          out = ws + WS_XQ + r; }
    else if (r < 5120)  { W = wk + (size_t)(r - 4096) * DIM; out = ws + WS_XK + (r - 4096); }
    else                { W = wv + (size_t)(r - 5120) * DIM; out = ws + WS_XV + (r - 5120); }

    const float4* w4 = (const float4*)W;
    const float4* x4 = (const float4*)x;
    float acc = 0.f;
#pragma unroll 8
    for (int i = 0; i < 32; ++i) {
        __builtin_prefetch(&w4[lane + 32 * ((i + 8) & 31)], 0, 0);
        float4 a = w4[lane + 32 * i];
        float4 b = x4[lane + 32 * i];
        acc += a.x * b.x + a.y * b.y + a.z * b.z + a.w * b.w;
    }
    for (int o = 16; o > 0; o >>= 1) acc += __shfl_xor(acc, o, 32);
    if (lane == 0) *out = acc;
}

// ---------------- flash-decode split: async-LDS double buffer + WMMA ----------------
__global__ __launch_bounds__(256) void attn_split(const float* __restrict__ cache_k,
                                                  const float* __restrict__ cache_v,
                                                  const float* __restrict__ cosv,
                                                  const float* __restrict__ sinv,
                                                  float* __restrict__ ws) {
    const int split = blockIdx.x;      // 0..NS-1
    const int h     = blockIdx.y;      // kv head 0..7
    const int t     = threadIdx.x;
    const int wave  = t >> 5, lane = t & 31;
    const int base  = split * CHUNK;

    __shared__ float    sBuf0[SUB * 128];   // 64KB K/V phase buffer (f32)
    __shared__ float    sBuf1[SUB * 128];   // 64KB K/V phase buffer (f32)
    __shared__ _Float16 sQ[16 * 128];       // Q tile (f16), rows 4..15 zero
    __shared__ float    sS[4 * CHUNK];      // raw scores
    __shared__ _Float16 sP[16 * CHUNK];     // softmax probs, rows 4..15 zero

    // --- issue one 128-key phase of async global->LDS copies (16 x b128/thread) ---
    auto issuePhase = [&](const float* src, int phase, float* buf) {
        const unsigned ldsBase = (unsigned)(uintptr_t)buf;
        const int basePos = base + phase * SUB;
#pragma unroll
        for (int i = 0; i < 16; ++i) {
            int slot = t + 256 * i;           // float4 slot 0..4095
            int pl = slot >> 5, d4 = slot & 31;
            const float* g = src + ((size_t)(basePos + pl) * N_KV_HEADS + h) * HEAD_DIM + d4 * 4;
            async_ld_b128(ldsBase + (unsigned)((pl * 128 + d4 * 4) * 4), g);
        }
    };

    // --- patch position START_POS (new token K with RoPE / new V) after wait ---
    auto patchPhase = [&](bool isK, int phase, float* buf) {
        const int basePos = base + phase * SUB;
        if (basePos <= START_POS && START_POS < basePos + SUB) {   // uniform
            if (t < 32) {
                const int row = START_POS - basePos;
                const int d0  = t * 4;
                if (isK) {
                    float a0 = ws[WS_XK + h * 128 + d0 + 0];
                    float a1 = ws[WS_XK + h * 128 + d0 + 1];
                    float a2 = ws[WS_XK + h * 128 + d0 + 2];
                    float a3 = ws[WS_XK + h * 128 + d0 + 3];
                    float c0 = cosv[(d0 >> 1)],     s0 = sinv[(d0 >> 1)];
                    float c1 = cosv[(d0 >> 1) + 1], s1 = sinv[(d0 >> 1) + 1];
                    buf[row * 128 + d0 + 0] = a0 * c0 - a1 * s0;
                    buf[row * 128 + d0 + 1] = a0 * s0 + a1 * c0;
                    buf[row * 128 + d0 + 2] = a2 * c1 - a3 * s1;
                    buf[row * 128 + d0 + 3] = a2 * s1 + a3 * c1;
                } else {
                    buf[row * 128 + d0 + 0] = ws[WS_XV + h * 128 + d0 + 0];
                    buf[row * 128 + d0 + 1] = ws[WS_XV + h * 128 + d0 + 1];
                    buf[row * 128 + d0 + 2] = ws[WS_XV + h * 128 + d0 + 2];
                    buf[row * 128 + d0 + 3] = ws[WS_XV + h * 128 + d0 + 3];
                }
            }
            __syncthreads();
        }
    };

    // --- prefetch K phases 0 and 1 into both buffers immediately ---
    issuePhase(cache_k, 0, sBuf0);
    issuePhase(cache_k, 1, sBuf1);

    // --- stage Q with RoPE while async copies fly ---
    {
        const int q = t >> 6, i = t & 63;                 // 256 pairs
        const float xr = ws[WS_XQ + (h * 4 + q) * 128 + 2 * i];
        const float xi = ws[WS_XQ + (h * 4 + q) * 128 + 2 * i + 1];
        const float c = cosv[i], s = sinv[i];
        sQ[q * 128 + 2 * i]     = (_Float16)(xr * c - xi * s);
        sQ[q * 128 + 2 * i + 1] = (_Float16)(xr * s + xi * c);
    }
    for (int i = t + 512; i < 16 * 128; i += 256)         sQ[i] = (_Float16)0.f;
    for (int i = t + 4 * CHUNK; i < 16 * CHUNK; i += 256) sP[i] = (_Float16)0.f;

    const int mrow = lane & 15;
    const int hi   = (lane < 16) ? 0 : 8;
    const int off  = (lane < 16) ? 0 : 16;

    // --- scores: each wave owns one 16-key tile per phase ---
    auto scoreTile = [&](int ph, const float* bufK) {
        const int krow0 = wave * 16;
        v8f c = {};
#pragma unroll
        for (int kk = 0; kk < 4; ++kk) {
            v16h a, b;
#pragma unroll
            for (int half = 0; half < 2; ++half)
#pragma unroll
                for (int v = 0; v < 4; ++v) {
                    int k = kk * 32 + off + half * 8 + 2 * v;   // A layout 16x32 f16, contiguous k = 16h + s
                    a[half * 8 + 2 * v]     = (_Float16)bufK[(krow0 + mrow) * 128 + k];
                    a[half * 8 + 2 * v + 1] = (_Float16)bufK[(krow0 + mrow) * 128 + k + 1];
                }
#pragma unroll
            for (int j = 0; j < 8; ++j) {                       // B layout 32x16 f16, contiguous k = 16h + s
                int k = kk * 32 + off + 2 * j;
                b[2 * j]     = sQ[mrow * 128 + k];
                b[2 * j + 1] = sQ[mrow * 128 + k + 1];
            }
            c = __builtin_amdgcn_wmma_f32_16x16x32_f16(false, a, false, b,
                                                       (short)0, c, false, false);
        }
        // C layout: VGPR r, lanes0-15 -> M=r, lanes16-31 -> M=r+8; col = lane%16
        if (mrow < 4) {
            const int rbase = (lane < 16) ? 0 : 8;
#pragma unroll
            for (int r = 0; r < 8; ++r)
                sS[mrow * CHUNK + ph * SUB + krow0 + rbase + r] = c[r] * SCALE;
        }
    };

    // --- P.V: each wave owns a 16-wide dim tile; C persists across phases ---
    v8f cacc = {};
    auto pvPhase = [&](int ph, const float* bufV) {
        const int wd = wave * 16;
#pragma unroll
        for (int lk = 0; lk < 4; ++lk) {
            v16h a, b;
#pragma unroll
            for (int half = 0; half < 2; ++half)
#pragma unroll
                for (int v = 0; v < 4; ++v) {
                    int kc = ph * SUB + lk * 32 + off + half * 8 + 2 * v;
                    a[half * 8 + 2 * v]     = sP[mrow * CHUNK + kc];
                    a[half * 8 + 2 * v + 1] = sP[mrow * CHUNK + kc + 1];
                }
#pragma unroll
            for (int j = 0; j < 8; ++j) {
                int kr = lk * 32 + off + 2 * j;
                b[2 * j]     = (_Float16)bufV[kr * 128 + wd + mrow];
                b[2 * j + 1] = (_Float16)bufV[(kr + 1) * 128 + wd + mrow];
            }
            cacc = __builtin_amdgcn_wmma_f32_16x16x32_f16(false, a, false, b,
                                                          (short)0, cacc, false, false);
        }
    };

    // ================= pipelined schedule =================
    WAIT_ASYNC(16);            // K0 landed (this wave)
    __syncthreads();           // block-wide: K0 + sQ/sP init visible
    patchPhase(true, 0, sBuf0);
    scoreTile(0, sBuf0);
    __syncthreads();           // all reads of sBuf0 done
    issuePhase(cache_v, 0, sBuf0);      // start V0 into freed buffer

    WAIT_ASYNC(16);            // K1 landed (outstanding: V0 only)
    __syncthreads();
    patchPhase(true, 1, sBuf1);
    scoreTile(1, sBuf1);
    __syncthreads();           // all reads of sBuf1 done; sS complete
    issuePhase(cache_v, 1, sBuf1);      // start V1

    // --- per-q softmax over the 256-key chunk (waves 0..3, one q each) ---
    if (wave < 4) {
        const int q = wave;
        float vals[8];
        float mx = -1e30f;
#pragma unroll
        for (int i = 0; i < 8; ++i) {
            vals[i] = sS[q * CHUNK + lane + 32 * i];
            mx = fmaxf(mx, vals[i]);
        }
        for (int o = 16; o > 0; o >>= 1) mx = fmaxf(mx, __shfl_xor(mx, o, 32));
        float l = 0.f;
#pragma unroll
        for (int i = 0; i < 8; ++i) {
            float p = __expf(vals[i] - mx);
            l += p;
            sP[q * CHUNK + lane + 32 * i] = (_Float16)p;
        }
        for (int o = 16; o > 0; o >>= 1) l += __shfl_xor(l, o, 32);
        if (lane == 0) {
            const int qh = h * 4 + q;
            ws[WS_M + qh * NS + split] = mx;
            ws[WS_L + qh * NS + split] = l;
        }
    }
    __syncthreads();           // sP ready

    WAIT_ASYNC(16);            // V0 landed (outstanding: V1)
    __syncthreads();
    patchPhase(false, 0, sBuf0);
    pvPhase(0, sBuf0);

    WAIT_ASYNC(0);             // V1 landed
    __syncthreads();
    patchPhase(false, 1, sBuf1);
    pvPhase(1, sBuf1);

    // valid C rows: q = r (0..3) on lanes 0..15; col d = wave*16 + lane
    if (lane < 16) {
#pragma unroll
        for (int r = 0; r < 4; ++r) {
            const int qh = h * 4 + r;
            const int d  = wave * 16 + lane;
            ws[WS_ACC + ((size_t)qh * NS + split) * 128 + d] = cacc[r];
        }
    }
}

// ---------------- merge split-K partials (log-sum-exp) ----------------
__global__ __launch_bounds__(128) void attn_reduce(float* __restrict__ ws) {
    const int qh = blockIdx.x;        // 0..31
    const int d  = threadIdx.x;       // 0..127
    float mstar = -1e30f;
    for (int s = 0; s < NS; ++s) mstar = fmaxf(mstar, ws[WS_M + qh * NS + s]);
    float num = 0.f, den = 0.f;
    for (int s = 0; s < NS; ++s) {
        float w = __expf(ws[WS_M + qh * NS + s] - mstar);
        den += w * ws[WS_L + qh * NS + s];
        num += w * ws[WS_ACC + ((size_t)qh * NS + s) * 128 + d];
    }
    ws[WS_ATTN + qh * 128 + d] = num / den;
}

// ---------------- output projection ----------------
__global__ __launch_bounds__(256) void oproj(const float* __restrict__ wo,
                                             const float* __restrict__ ws,
                                             float* __restrict__ out) {
    const int wave = threadIdx.x >> 5, lane = threadIdx.x & 31;
    const int r = blockIdx.x * 8 + wave;          // 0..4095
    const float4* w4 = (const float4*)(wo + (size_t)r * DIM);
    const float4* a4 = (const float4*)(ws + WS_ATTN);
    float acc = 0.f;
#pragma unroll 8
    for (int i = 0; i < 32; ++i) {
        __builtin_prefetch(&w4[lane + 32 * ((i + 8) & 31)], 0, 0);
        float4 a = w4[lane + 32 * i];
        float4 b = a4[lane + 32 * i];
        acc += a.x * b.x + a.y * b.y + a.z * b.z + a.w * b.w;
    }
    for (int o = 16; o > 0; o >>= 1) acc += __shfl_xor(acc, o, 32);
    if (lane == 0) out[r] = acc;
}

extern "C" void kernel_launch(void* const* d_in, const int* in_sizes, int n_in,
                              void* d_out, int out_size, void* d_ws, size_t ws_size,
                              hipStream_t stream) {
    const float* x       = (const float*)d_in[0];
    const float* wq      = (const float*)d_in[1];
    const float* wk      = (const float*)d_in[2];
    const float* wv      = (const float*)d_in[3];
    const float* wo      = (const float*)d_in[4];
    const float* cache_k = (const float*)d_in[5];
    const float* cache_v = (const float*)d_in[6];
    const float* cosv    = (const float*)d_in[7];
    const float* sinv    = (const float*)d_in[8];
    float* out = (float*)d_out;
    float* ws  = (float*)d_ws;

    qkv_gemv<<<768, 256, 0, stream>>>(x, wq, wk, wv, ws);
    attn_split<<<dim3(NS, N_KV_HEADS), 256, 0, stream>>>(cache_k, cache_v, cosv, sinv, ws);
    attn_reduce<<<N_HEADS, HEAD_DIM, 0, stream>>>(ws);
    oproj<<<512, 256, 0, stream>>>(wo, ws, out);
}
